// RoPEAttention_74921409512073
// MI455X (gfx1250) — matrix operand order repaired and checked by reading, not compiled, not run on hardware
//
#include <hip/hip_runtime.h>
#include <hip/hip_bf16.h>

// Shapes fixed by the reference: B=2, T=2048, C=2048, H=16, D=128.
#define BB   2
#define TSEQ 2048
#define CH   2048
#define NH   16
#define HD   128

typedef __attribute__((ext_vector_type(16))) __bf16 v16bf;
typedef __attribute__((ext_vector_type(8)))  __bf16 v8bf;
typedef __attribute__((ext_vector_type(4)))  __bf16 v4bf;
typedef __attribute__((ext_vector_type(8)))  float  v8f;

__device__ __forceinline__ __bf16 f2bf(float x) { return (__bf16)x; }

__device__ __forceinline__ v4bf pack4(float a, float b, float c, float d) {
    v4bf r; r[0] = (__bf16)a; r[1] = (__bf16)b; r[2] = (__bf16)c; r[3] = (__bf16)d;
    return r;
}

// Two contiguous 16-byte LDS loads -> one 16-element fragment (ds_load_b128 x2)
__device__ __forceinline__ v16bf ld_frag(const __bf16* p0, const __bf16* p1) {
    v8bf lo = *(const v8bf*)p0;
    v8bf hi = *(const v8bf*)p1;
    return __builtin_shufflevector(lo, hi, 0, 1, 2, 3, 4, 5, 6, 7,
                                           8, 9, 10, 11, 12, 13, 14, 15);
}

// ---------------------------------------------------------------------------
// GEMM: C[4096 x 2048] = A[fp32,lda] x B[fp32,ldb], bf16 WMMA, f32 accum.
// Block: 256 threads (8 waves). Block tile 128(M) x 64(N). K-chunk 32.
// mode 0: row-major store; 1: (B,H,T,D) scatter; 2: scatter + fused RoPE.
// ---------------------------------------------------------------------------
__global__ __launch_bounds__(256) void gemm_bf16_wmma(
    const float* __restrict__ A, const float* __restrict__ Bm,
    float* __restrict__ Cout, int lda, int ldb, int mode)
{
    __shared__ __bf16 sA[128][32];   // row-major: lane A-frag = 2 contiguous runs
    __shared__ __bf16 sBT[64][32];   // B transposed: [n][k] -> contiguous K runs

    const int tid  = threadIdx.x;
    const int wave = tid >> 5;
    const int lane = tid & 31;
    const int g    = lane >> 4;
    const int l15  = lane & 15;
    const int rowBase = blockIdx.y * 128;
    const int colBase = blockIdx.x * 64;

    // Precomputed staging coordinates (constant across K loop)
    const int ar[4] = { (tid + 0)   >> 3, (tid + 256) >> 3,
                        (tid + 512) >> 3, (tid + 768) >> 3 };
    const int ac = (tid & 7) * 4;
    const int br[2] = { tid >> 4, (tid + 256) >> 4 };
    const int bc = (tid & 15) * 4;

    v8f acc[4];
#pragma unroll
    for (int nt = 0; nt < 4; ++nt) acc[nt] = {};

    for (int kk = 0; kk < CH; kk += 32) {
        if (kk + 32 < CH) {  // prefetch next chunks (global_prefetch_b8)
            __builtin_prefetch(&A[(size_t)(rowBase + (tid >> 1)) * lda + kk + 32], 0, 1);
            __builtin_prefetch(&Bm[(size_t)(kk + 32 + (tid >> 3)) * ldb + colBase], 0, 1);
        }
        // ---- staging: batch all global loads first (clause), then store ----
        float4 fa[4], fb[2];
#pragma unroll
        for (int t = 0; t < 4; ++t)
            fa[t] = *(const float4*)&A[(size_t)(rowBase + ar[t]) * lda + kk + ac];
#pragma unroll
        for (int t = 0; t < 2; ++t)
            fb[t] = *(const float4*)&Bm[(size_t)(kk + br[t]) * ldb + colBase + bc];
#pragma unroll
        for (int t = 0; t < 4; ++t)
            *(v4bf*)&sA[ar[t]][ac] = pack4(fa[t].x, fa[t].y, fa[t].z, fa[t].w);
#pragma unroll
        for (int t = 0; t < 2; ++t) {
            sBT[bc + 0][br[t]] = f2bf(fb[t].x);
            sBT[bc + 1][br[t]] = f2bf(fb[t].y);
            sBT[bc + 2][br[t]] = f2bf(fb[t].z);
            sBT[bc + 3][br[t]] = f2bf(fb[t].w);
        }
        __syncthreads();

        // ---- load all fragments, then issue 4 WMMAs back-to-back ----
        const __bf16* arow = &sA[wave * 16 + l15][0];
        v16bf a = ld_frag(arow + 8 * g, arow + 16 + 8 * g);
        v16bf bfr[4];
#pragma unroll
        for (int nt = 0; nt < 4; ++nt) {
            const __bf16* brow = &sBT[nt * 16 + l15][0];
            bfr[nt] = ld_frag(brow + 8 * g, brow + 16 + 8 * g);
        }
#pragma unroll
        for (int nt = 0; nt < 4; ++nt)
            acc[nt] = __builtin_amdgcn_wmma_f32_16x16x32_bf16(
                false, a, false, bfr[nt], (short)0, acc[nt], false, false);
        __syncthreads();
    }

    // Epilogue. D-layout: lane l15 = N, VGPR r -> M = r + 8g.
#pragma unroll
    for (int nt = 0; nt < 4; ++nt) {
#pragma unroll
        for (int r = 0; r < 8; ++r) {
            int m   = r + 8 * g;
            int row = rowBase + wave * 16 + m;        // bt
            int col = colBase + nt * 16 + l15;        // channel
            float val = acc[nt][r];
            if (mode == 0) {
                Cout[(size_t)row * CH + col] = val;
            } else {
                if (mode == 2) {
                    float partner = __shfl_xor(val, 1, 32);
                    int d = col & (HD - 1);
                    int j = d >> 1;
                    int t = row & (TSEQ - 1);
                    float freq = (float)t * __powf(10000.f, -(float)(2 * j) / (float)HD);
                    float sn, cs;
                    __sincosf(freq, &sn, &cs);
                    val = (d & 1) ? fmaf(partner, sn, val * cs)
                                  : fmaf(val, cs, -partner * sn);
                }
                int bIdx = row >> 11;
                int t    = row & (TSEQ - 1);
                int h    = col >> 7;
                int d    = col & (HD - 1);
                Cout[(((size_t)bIdx * NH + h) * TSEQ + t) * HD + d] = val;
            }
        }
    }
}

// ---------------------------------------------------------------------------
// Flash attention: grid (T/64, H, B), block 128 (4 waves).
// Wave: 16 query rows, online softmax over T keys, 32 keys/step.
// ---------------------------------------------------------------------------
__global__ __launch_bounds__(128) void attn_flash_wmma(
    const float* __restrict__ q, const float* __restrict__ k,
    const float* __restrict__ v, float* __restrict__ o)
{
    __shared__ __bf16 sK[32][HD];      // row-major: B-frag K runs contiguous in d
    __shared__ __bf16 sVT[HD][32];     // V transposed: [d][key]
    __shared__ __bf16 sP[4][16][32];   // per-wave P relayout bounce

    const int tid  = threadIdx.x;
    const int wave = tid >> 5;
    const int lane = tid & 31;
    const int g    = lane >> 4;
    const int l15  = lane & 15;
    const int b    = blockIdx.z;
    const int h    = blockIdx.y;
    const int qRow = blockIdx.x * 64 + wave * 16;

    const size_t headOff = ((size_t)b * NH + h) * (size_t)TSEQ * HD;
    const float* qh = q + headOff;
    const float* kh = k + headOff;
    const float* vh = v + headOff;

    // Q tile 16x128 as 4 A-fragments
    v16bf qa[4];
#pragma unroll
    for (int c4 = 0; c4 < 4; ++c4) {
        const float* qp = &qh[(size_t)(qRow + l15) * HD + c4 * 32];
        float4 x0 = *(const float4*)(qp + 8 * g);
        float4 x1 = *(const float4*)(qp + 8 * g + 4);
        float4 x2 = *(const float4*)(qp + 16 + 8 * g);
        float4 x3 = *(const float4*)(qp + 16 + 8 * g + 4);
        qa[c4][0]  = f2bf(x0.x); qa[c4][1]  = f2bf(x0.y);
        qa[c4][2]  = f2bf(x0.z); qa[c4][3]  = f2bf(x0.w);
        qa[c4][4]  = f2bf(x1.x); qa[c4][5]  = f2bf(x1.y);
        qa[c4][6]  = f2bf(x1.z); qa[c4][7]  = f2bf(x1.w);
        qa[c4][8]  = f2bf(x2.x); qa[c4][9]  = f2bf(x2.y);
        qa[c4][10] = f2bf(x2.z); qa[c4][11] = f2bf(x2.w);
        qa[c4][12] = f2bf(x3.x); qa[c4][13] = f2bf(x3.y);
        qa[c4][14] = f2bf(x3.z); qa[c4][15] = f2bf(x3.w);
    }

    v8f O[8];
#pragma unroll
    for (int c = 0; c < 8; ++c) O[c] = {};
    float mrow[8], lrow[8];
#pragma unroll
    for (int r = 0; r < 8; ++r) { mrow[r] = -1e30f; lrow[r] = 0.f; }

    const float scale = 0.0883883476483184f;  // 1/sqrt(128)

    // Staging coords: thread handles 8 float4s (key rr, d cc)
    const int cc = (tid & 31) * 4;

    for (int kb = 0; kb < TSEQ; kb += 32) {
        __syncthreads();
        // batch loads (4 at a time), then stores
#pragma unroll
        for (int half = 0; half < 2; ++half) {
            float4 fk[4], fv[4];
#pragma unroll
            for (int t = 0; t < 4; ++t) {
                int rr = (tid + (half * 4 + t) * 128) >> 5;
                size_t gi = (size_t)(kb + rr) * HD + cc;
                fk[t] = *(const float4*)&kh[gi];
                fv[t] = *(const float4*)&vh[gi];
            }
#pragma unroll
            for (int t = 0; t < 4; ++t) {
                int rr = (tid + (half * 4 + t) * 128) >> 5;
                *(v4bf*)&sK[rr][cc] = pack4(fk[t].x, fk[t].y, fk[t].z, fk[t].w);
                sVT[cc + 0][rr] = f2bf(fv[t].x);
                sVT[cc + 1][rr] = f2bf(fv[t].y);
                sVT[cc + 2][rr] = f2bf(fv[t].z);
                sVT[cc + 3][rr] = f2bf(fv[t].w);
            }
        }
        if (kb + 32 < TSEQ) {
            __builtin_prefetch(&kh[(size_t)(kb + 32 + (tid >> 2)) * HD], 0, 1);
            __builtin_prefetch(&vh[(size_t)(kb + 32 + (tid >> 2)) * HD], 0, 1);
        }
        __syncthreads();

        // scores: S[16q x 32keys] = Q @ K^T; load 4 frags, then 4 WMMAs
        v8f s0 = {}, s1 = {};
        const __bf16* k0 = &sK[l15][0];
        const __bf16* k1 = &sK[16 + l15][0];
#pragma unroll
        for (int cp = 0; cp < 2; ++cp) {
            v16bf bk[4];
#pragma unroll
            for (int t = 0; t < 2; ++t) {
                int off = (cp * 2 + t) * 32 + 8 * g;
                bk[t * 2 + 0] = ld_frag(k0 + off, k0 + off + 16);
                bk[t * 2 + 1] = ld_frag(k1 + off, k1 + off + 16);
            }
            s0 = __builtin_amdgcn_wmma_f32_16x16x32_bf16(
                false, qa[cp * 2 + 0], false, bk[0], (short)0, s0, false, false);
            s1 = __builtin_amdgcn_wmma_f32_16x16x32_bf16(
                false, qa[cp * 2 + 0], false, bk[1], (short)0, s1, false, false);
            s0 = __builtin_amdgcn_wmma_f32_16x16x32_bf16(
                false, qa[cp * 2 + 1], false, bk[2], (short)0, s0, false, false);
            s1 = __builtin_amdgcn_wmma_f32_16x16x32_bf16(
                false, qa[cp * 2 + 1], false, bk[3], (short)0, s1, false, false);
        }

        // online softmax; rows m = r + 8g live on the 16 lanes of group g
#pragma unroll
        for (int r = 0; r < 8; ++r) {
            float a0 = s0[r] * scale, a1 = s1[r] * scale;
            float mx = fmaxf(a0, a1);
            mx = fmaxf(mx, __shfl_xor(mx, 1, 32));
            mx = fmaxf(mx, __shfl_xor(mx, 2, 32));
            mx = fmaxf(mx, __shfl_xor(mx, 4, 32));
            mx = fmaxf(mx, __shfl_xor(mx, 8, 32));
            float mnew  = fmaxf(mrow[r], mx);
            float alpha = __expf(mrow[r] - mnew);
            float p0 = __expf(a0 - mnew);
            float p1 = __expf(a1 - mnew);
            float ps = p0 + p1;
            ps += __shfl_xor(ps, 1, 32);
            ps += __shfl_xor(ps, 2, 32);
            ps += __shfl_xor(ps, 4, 32);
            ps += __shfl_xor(ps, 8, 32);
            lrow[r] = lrow[r] * alpha + ps;
            mrow[r] = mnew;
#pragma unroll
            for (int c = 0; c < 8; ++c) O[c][r] *= alpha;
            sP[wave][r + 8 * g][l15]      = f2bf(p0);
            sP[wave][r + 8 * g][16 + l15] = f2bf(p1);
        }
        asm volatile("s_wait_dscnt 0" ::: "memory");  // wave-local LDS RAW

        const __bf16* pr = &sP[wave][l15][0];
        v16bf pa = ld_frag(pr + 8 * g, pr + 16 + 8 * g);

        // O += P(16x32) @ V(32x128); 4 frags loaded, then 4 WMMAs, twice
#pragma unroll
        for (int cp = 0; cp < 2; ++cp) {
            v16bf bv[4];
#pragma unroll
            for (int t = 0; t < 4; ++t) {
                const __bf16* vr = &sVT[(cp * 4 + t) * 16 + l15][0];
                bv[t] = ld_frag(vr + 8 * g, vr + 16 + 8 * g);
            }
#pragma unroll
            for (int t = 0; t < 4; ++t)
                O[cp * 4 + t] = __builtin_amdgcn_wmma_f32_16x16x32_bf16(
                    false, pa, false, bv[t], (short)0, O[cp * 4 + t], false, false);
        }
    }

    // normalize and write (B,T,C)
#pragma unroll
    for (int c = 0; c < 8; ++c)
#pragma unroll
        for (int r = 0; r < 8; ++r) {
            int m = r + 8 * g;
            int t = qRow + m;
            float val = O[c][r] / lrow[r];
            o[((size_t)b * TSEQ + t) * CH + h * HD + c * 16 + l15] = val;
        }
}

// ---------------------------------------------------------------------------
extern "C" void kernel_launch(void* const* d_in, const int* in_sizes, int n_in,
                              void* d_out, int out_size, void* d_ws, size_t ws_size,
                              hipStream_t stream) {
    (void)in_sizes; (void)n_in; (void)out_size; (void)ws_size;
    const float* x    = (const float*)d_in[0];
    const float* W_qk = (const float*)d_in[1];
    const float* W_v  = (const float*)d_in[2];
    const float* W_o  = (const float*)d_in[3];

    const size_t BTC  = (size_t)BB * TSEQ * CH;
    const size_t BHTD = (size_t)BB * NH * TSEQ * HD;

    float* out  = (float*)d_out;          // (B,T,C)
    float* kout = out + BTC;              // (B,H,T,D) — returned output
    float* vout = kout + BHTD;            // (B,H,T,D) — returned output
    float* qws  = (float*)d_ws;           // (B,H,T,D) scratch
    float* aws  = qws + BHTD;             // (B,T,C)  scratch

    dim3 gg(CH / 64, (BB * TSEQ) / 128);
    gemm_bf16_wmma<<<gg, 256, 0, stream>>>(x, W_qk,      qws,  CH, 2 * CH, 2);
    gemm_bf16_wmma<<<gg, 256, 0, stream>>>(x, W_qk + CH, kout, CH, 2 * CH, 2);
    gemm_bf16_wmma<<<gg, 256, 0, stream>>>(x, W_v,       vout, CH, CH, 1);

    dim3 ga(TSEQ / 64, NH, BB);
    attn_flash_wmma<<<ga, 128, 0, stream>>>(qws, kout, vout, aws);

    gemm_bf16_wmma<<<gg, 256, 0, stream>>>(aws, W_o, out, CH, CH, 0);
}
